// VectorQuantizer_62225486185149
// MI455X (gfx1250) — matrix operand order repaired and checked by reading, not compiled, not run on hardware
//
#include <hip/hip_runtime.h>
#include <hip/hip_bf16.h>
#include <math.h>

typedef __attribute__((ext_vector_type(16))) __bf16 v16bf;
typedef __attribute__((ext_vector_type(8)))  float  v8f;

#define N_E    1024
#define E_DIM  512
#define B_SZ   32
#define T_SZ   2048
#define NROWS  (B_SZ * T_SZ)      /* 65536 */
#define M_TILE 128
#define NWG    (NROWS / M_TILE)   /* 512  */

/* workspace byte offsets */
#define WS_EBF_HI 0u
#define WS_EBF_LO (WS_EBF_HI + (unsigned)(N_E * E_DIM * 2))   /* 1 MiB */
#define WS_ENORM  (WS_EBF_LO + (unsigned)(N_E * E_DIM * 2))   /* 2 MiB */
#define WS_IDX    (WS_ENORM + (unsigned)(N_E * 4))
#define WS_HIST   (WS_IDX   + (unsigned)(NROWS * 4))
#define WS_DSUM   (WS_HIST  + (unsigned)(N_E * 4))

// ---------------------------------------------------------------------------
// Async memory->LDS copy of 32 contiguous bytes via the CDNA5 Tensor/async
// path (GLOBAL_LOAD_ASYNC_TO_LDS_B128, tracked with ASYNCcnt; INST_OFFSET is
// added to BOTH the LDS and global addresses per ISA 15.18).
// ---------------------------------------------------------------------------
__device__ __forceinline__ void async_copy32(const __bf16* src, __bf16* dst) {
  unsigned lds = (unsigned)(size_t)(void*)dst;            // low 32 bits = LDS offset
  unsigned long long ga = (unsigned long long)(size_t)(const void*)src;
  asm volatile("global_load_async_to_lds_b128 %0, %1, off"
               :: "v"(lds), "v"(ga) : "memory");
  asm volatile("global_load_async_to_lds_b128 %0, %1, off offset:16"
               :: "v"(lds), "v"(ga) : "memory");
}
__device__ __forceinline__ void wait_async0() {
  asm volatile("s_wait_asynccnt 0x0" ::: "memory");
}

// ---------------------------------------------------------------------------
// Prep: split emb into bf16 hi/lo, pre-swizzled into WMMA B-fragment layout
// [jtile(64)][ktile(16)][lane(32)][e(16)]; also ||e_j||^2, zero hist/dsum.
// B-frag (32x16 bf16): lane L holds column n=L&15, K = (L>>4)*16 + e.
// ---------------------------------------------------------------------------
__global__ void vq_prep(const float* __restrict__ emb,
                        __bf16* __restrict__ ehi, __bf16* __restrict__ elo,
                        float* __restrict__ enorm, int* __restrict__ hist,
                        float* __restrict__ dsum) {
  const int j = blockIdx.x;     // code index 0..1023
  const int t = threadIdx.x;    // 128 threads
  __shared__ float red[128];
  float s = 0.f;
  for (int k = 0; k < 4; ++k) {
    int d = t + k * 128;
    float v = emb[j * E_DIM + d];
    __bf16 h = (__bf16)v;
    __bf16 l = (__bf16)(v - (float)h);
    int kt = d >> 5, kk = d & 31;
    int lane = ((kk >> 4) << 4) | (j & 15);
    int e = kk & 15;
    int off = (((j >> 4) * 16 + kt) * 32 + lane) * 16 + e;
    ehi[off] = h;
    elo[off] = l;
    s += v * v;
  }
  red[t] = s;
  __syncthreads();
  for (int w = 64; w > 0; w >>= 1) {
    if (t < w) red[t] += red[t + w];
    __syncthreads();
  }
  if (t == 0) {
    enorm[j] = red[0];
    hist[j] = 0;
    if (j == 0) *dsum = 0.f;
  }
}

// ---------------------------------------------------------------------------
// Main: per-WG 128 rows x all 1024 codes. Split-bf16 WMMA (hh + hl + lh)
// accumulated in fp32, double-buffered async B staging (1 barrier / k-step),
// argmin with index tie-break, histogram + loss partials.
// A-frag (16x32 bf16): lane L holds row m=L&15, K = (L>>4)*16 + e.
// C/D: lane L holds column n=L&15; VGPR r holds row (L>>4)*8 + r.
// ---------------------------------------------------------------------------
__global__ __launch_bounds__(256, 1)
void vq_main(const float* __restrict__ z,
             const __bf16* __restrict__ ehi, const __bf16* __restrict__ elo,
             const float* __restrict__ enorm,
             int* __restrict__ idxout, float* __restrict__ idxf,
             int* __restrict__ hist, float* __restrict__ dsum) {
  __shared__ __bf16 sAhi[16 * 8 * 32 * 16];   // 128 KiB  [kt][mt][lane][e]
  __shared__ __bf16 sAlo[16 * 8 * 32 * 16];   // 128 KiB
  __shared__ __bf16 sBhi[2][4 * 32 * 16];     // 2 x 4 KiB [jt][lane][e]
  __shared__ __bf16 sBlo[2][4 * 32 * 16];     // 2 x 4 KiB
  __shared__ float znpart[256];
  __shared__ float rowdmin[128];
  __shared__ float sAcc;

  const int tid = threadIdx.x;
  const int wg  = blockIdx.x;
  const int b   = wg >> 4;                // 16 tiles per batch (2048/128)
  const int t0  = (wg & 15) * M_TILE;
  const int n0  = wg * M_TILE;

  if (tid == 0) sAcc = 0.f;

  // ---- stage A (z tile) into LDS in fragment layout; coalesced along t ----
  float zn = 0.f;
  const float* zb = z + ((size_t)b * E_DIM) * T_SZ + t0;
  for (int s = 0; s < 256; ++s) {
    int i = tid + (s << 8);
    int d = i >> 7, m = i & 127;          // m = tid&127 constant per thread
    float v = zb[(size_t)d * T_SZ + m];
    __bf16 h = (__bf16)v;
    __bf16 l = (__bf16)(v - (float)h);
    int kt = d >> 5, kk = d & 31;
    int hbit = kk >> 4;
    int e = kk & 15;
    int lane = (hbit << 4) | (m & 15);
    int off = (((kt << 3) + (m >> 4)) * 32 + lane) * 16 + e;
    sAhi[off] = h;
    sAlo[off] = l;
    zn += v * v;
  }
  znpart[tid] = zn;

  const int L  = tid & 31;
  const int mt = tid >> 5;                // wave id == row-subtile
  const int jl = L & 15;

  float bd[8];
  int   bj[8];
  #pragma unroll
  for (int r = 0; r < 8; ++r) { bd[r] = INFINITY; bj[r] = 0; }

  // cooperative async B-copy mapping for this thread (32 B each, 8 KiB/step)
  const int carr = tid >> 7;              // 0 = hi, 1 = lo
  const int cjt  = (tid >> 5) & 3;
  const int cc   = tid & 31;
  const __bf16* csrc0 = (carr ? elo : ehi) + (cjt * 16 * 512) + cc * 16;
  __bf16* cdst0 = (carr ? sBlo[0] : sBhi[0]) + (cjt * 32 + cc) * 16;
  __bf16* cdst1 = (carr ? sBlo[1] : sBhi[1]) + (cjt * 32 + cc) * 16;

  // prologue: (jb=0, kt=0) -> buffer 0; also publishes staged A
  async_copy32(csrc0, cdst0);
  wait_async0();
  __syncthreads();

  for (int jb = 0; jb < 16; ++jb) {       // 64-code blocks
    v8f acc0 = {}, acc1 = {}, acc2 = {}, acc3 = {};
    float en0 = enorm[jb * 64 +  0 + jl];
    float en1 = enorm[jb * 64 + 16 + jl];
    float en2 = enorm[jb * 64 + 32 + jl];
    float en3 = enorm[jb * 64 + 48 + jl];

    for (int kt = 0; kt < 16; ++kt) {     // K = 512 in steps of 32
      const int cur = kt & 1;
      // prefetch next B tile (double-buffered; crosses jb boundary too)
      if (kt < 15) {
        async_copy32(csrc0 + jb * 32768 + (kt + 1) * 512, cur ? cdst0 : cdst1);
      } else if (jb < 15) {
        async_copy32(csrc0 + (jb + 1) * 32768, cur ? cdst0 : cdst1);
      }

      const __bf16* bufh = sBhi[cur];
      const __bf16* bufl = sBlo[cur];
      v16bf ahi = *(const v16bf*)&sAhi[((kt * 8 + mt) * 32 + L) * 16];
      v16bf alo = *(const v16bf*)&sAlo[((kt * 8 + mt) * 32 + L) * 16];
      #pragma unroll
      for (int jt = 0; jt < 4; ++jt) {
        v16bf bhi = *(const v16bf*)&bufh[(jt * 32 + L) * 16];
        v16bf blo = *(const v16bf*)&bufl[(jt * 32 + L) * 16];
        v8f& acc = (jt == 0) ? acc0 : (jt == 1) ? acc1 : (jt == 2) ? acc2 : acc3;
        acc = __builtin_amdgcn_wmma_f32_16x16x32_bf16(false, ahi, false, bhi,
                                                      (short)0, acc, false, false);
        acc = __builtin_amdgcn_wmma_f32_16x16x32_bf16(false, ahi, false, blo,
                                                      (short)0, acc, false, false);
        acc = __builtin_amdgcn_wmma_f32_16x16x32_bf16(false, alo, false, bhi,
                                                      (short)0, acc, false, false);
      }
      wait_async0();      // my writes into the other buffer have landed
      __syncthreads();    // publish; also retires this step's readers
    }

    // running argmin over this lane's j-subset (j increases monotonically)
    #pragma unroll
    for (int jt = 0; jt < 4; ++jt) {
      const v8f& acc = (jt == 0) ? acc0 : (jt == 1) ? acc1 : (jt == 2) ? acc2 : acc3;
      float en = (jt == 0) ? en0 : (jt == 1) ? en1 : (jt == 2) ? en2 : en3;
      int j = jb * 64 + jt * 16 + jl;
      #pragma unroll
      for (int r = 0; r < 8; ++r) {
        float dv = en - 2.0f * acc[r];
        if (dv < bd[r]) { bd[r] = dv; bj[r] = j; }
      }
    }
  }

  // cross-lane argmin within each 16-lane group (tie -> smallest j)
  #pragma unroll
  for (int r = 0; r < 8; ++r) {
    float d = bd[r];
    int   j = bj[r];
    #pragma unroll
    for (int off = 8; off >= 1; off >>= 1) {
      float od = __shfl_xor(d, off, 16);
      int   oj = __shfl_xor(j, off, 16);
      if (od < d || (od == d && oj < j)) { d = od; j = oj; }
    }
    if (jl == 0) {                        // lanes 0 and 16
      int mrow = mt * 16 + (L >> 4) * 8 + r;
      int n = n0 + mrow;
      idxout[n] = j;
      idxf[n] = (float)j;
      atomicAdd(&hist[j], 1);
      rowdmin[mrow] = d;                  // ||e||^2 - 2 z.e
    }
  }
  __syncthreads();
  if (tid < 128) {
    float s = rowdmin[tid] + znpart[tid] + znpart[tid + 128];  // ||z - e||^2
    atomicAdd(&sAcc, s);
  }
  __syncthreads();
  if (tid == 0) atomicAdd(dsum, sAcc);
}

// ---------------------------------------------------------------------------
// z_q gather: z_q[b,d,t] = emb[idx[b*T+t], d]; coalesced float4 writes,
// scattered reads stay in L2 (emb is 2 MB).
// ---------------------------------------------------------------------------
__global__ void vq_zq(const float* __restrict__ emb, const int* __restrict__ idx,
                      float* __restrict__ zq) {
  int i = blockIdx.x * blockDim.x + threadIdx.x;   // 8388608 total
  int t4 = i & 511;
  int d  = (i >> 9) & 511;
  int b  = i >> 18;
  int n  = b * T_SZ + t4 * 4;
  int4 id = *(const int4*)&idx[n];
  float4 o;
  o.x = emb[(size_t)id.x * E_DIM + d];
  o.y = emb[(size_t)id.y * E_DIM + d];
  o.z = emb[(size_t)id.z * E_DIM + d];
  o.w = emb[(size_t)id.w * E_DIM + d];
  *(float4*)&zq[((size_t)(b * E_DIM + d)) * T_SZ + t4 * 4] = o;
}

// ---------------------------------------------------------------------------
// Finalize: perplexity from histogram + loss scalar.
// ---------------------------------------------------------------------------
__global__ void vq_final(const int* __restrict__ hist, const float* __restrict__ dsum,
                         float* __restrict__ loss_out, float* __restrict__ perp_out) {
  __shared__ float red[256];
  int t = threadIdx.x;
  float s = 0.f;
  for (int k = 0; k < 4; ++k) {
    int j = t + k * 256;
    float p = (float)hist[j] * (1.0f / 65536.0f);
    s += p * logf(p + 1e-10f);
  }
  red[t] = s;
  __syncthreads();
  for (int w = 128; w > 0; w >>= 1) {
    if (t < w) red[t] += red[t + w];
    __syncthreads();
  }
  if (t == 0) {
    *perp_out = expf(-red[0]);
    *loss_out = 1.25f * (*dsum) / 33554432.0f;  // (1 + beta) * mean
  }
}

// ---------------------------------------------------------------------------
extern "C" void kernel_launch(void* const* d_in, const int* in_sizes, int n_in,
                              void* d_out, int out_size, void* d_ws, size_t ws_size,
                              hipStream_t stream) {
  const float* z   = (const float*)d_in[0];
  const float* emb = (const float*)d_in[1];
  char* ws = (char*)d_ws;
  __bf16* ehi  = (__bf16*)(ws + WS_EBF_HI);
  __bf16* elo  = (__bf16*)(ws + WS_EBF_LO);
  float* enorm = (float*)(ws + WS_ENORM);
  int*   idx   = (int*)(ws + WS_IDX);
  int*   hist  = (int*)(ws + WS_HIST);
  float* dsum  = (float*)(ws + WS_DSUM);

  float* out  = (float*)d_out;
  float* loss = out;                         // [0]
  float* zq   = out + 1;                     // [1 .. 33554432]
  float* perp = out + 1 + 33554432;          // [33554433]
  float* idxf = out + 2 + 33554432;          // [33554434 ..]

  vq_prep <<<N_E, 128, 0, stream>>>(emb, ehi, elo, enorm, hist, dsum);
  vq_main <<<NWG, 256, 0, stream>>>(z, ehi, elo, enorm, idx, idxf, hist, dsum);
  vq_zq   <<<NROWS * E_DIM / 4 / 256, 256, 0, stream>>>(emb, idx, zq); // 32768 blocks
  vq_final<<<1, 256, 0, stream>>>(hist, dsum, loss, perp);
}
